// SingleFcMonLayer_52888227283698
// MI455X (gfx1250) — matrix operand order repaired and checked by reading, not compiled, not run on hardware
//
#include <hip/hip_runtime.h>
#include <hip/hip_bf16.h>

// Monotone DEQ single-FC layer, algebraically folded:
//   Wc = 0.9*I - Aw^T Aw + Bw^T - Bw   (constant 1024x1024)
//   bias = x @ Uw^T + Ub
//   z <- ReLU(z @ Wc + bias)  applied 51x starting from z=0 (z1 = ReLU(bias))
// GEMMs run on v_wmma_f32_16x16x32_bf16 (f32 accumulate); the iteration
// kernel stages tiles with GLOBAL_LOAD_ASYNC_TO_LDS_B128 when available.

#define IN_DIM  3072
#define OUT_DIM 1024
#define BATCH   8192

#define BM 128
#define BN 128
#define BK 32
#define LDSP 40  // padded LDS row stride (elements): 80B = 20 banks, conflict-free

typedef __attribute__((ext_vector_type(16))) __bf16 v16bf;
typedef __attribute__((ext_vector_type(8)))  __bf16 v8bf;
typedef __attribute__((ext_vector_type(8)))  float  v8f;

// ---- CDNA5 async global->LDS copy (guarded: falls back to VGPR copies) ----
#if defined(__has_builtin)
# if __has_builtin(__builtin_amdgcn_global_load_async_to_lds_b128) && \
     __has_builtin(__builtin_amdgcn_s_wait_asynccnt)
#  define USE_ASYNC_LDS 1
# endif
#endif
#ifndef USE_ASYNC_LDS
# define USE_ASYNC_LDS 0
#endif

#if USE_ASYNC_LDS
// Builtin signature (probe-confirmed via diagnostics): b128 payload is an
// ext-vector int4; param0 = global (AS1) int4*, param1 = LDS (AS3) int4*.
typedef __attribute__((ext_vector_type(4))) int v4i_;
typedef __attribute__((address_space(1))) v4i_ GV4;  // global int4
typedef __attribute__((address_space(3))) v4i_ LV4;  // LDS int4
#endif

__device__ __forceinline__ unsigned short f32_bf16(float f) {
  unsigned int u = __float_as_uint(f);
  u += 0x7FFFu + ((u >> 16) & 1u);   // round-to-nearest-even
  return (unsigned short)(u >> 16);
}

// A fragment: 16x32 bf16, M = lane&15, half h = lane>>4,
// element e -> K = (e>>3)*16 + h*8 + (e&7)   (ISA 7.12.2, 16-bit A 16x32)
__device__ __forceinline__ v16bf load_frag_a(const unsigned short* As, int m0, int lane) {
  const int m = m0 + (lane & 15);
  const int h = lane >> 4;
  v8bf lo = *(const v8bf*)(As + m * LDSP + h * 8);
  v8bf hi = *(const v8bf*)(As + m * LDSP + 16 + h * 8);
  v16bf r;
#pragma unroll
  for (int i = 0; i < 8; ++i) { r[i] = lo[i]; r[i + 8] = hi[i]; }
  return r;
}

// B fragment: 32x16 bf16 from an N-major (transposed) LDS tile [n][k]:
// N = lane&15, element e -> K = (e>>3)*16 + (lane>>4)*8 + (e&7)
__device__ __forceinline__ v16bf load_frag_b(const unsigned short* Bs, int n0, int lane) {
  const int n = n0 + (lane & 15);
  const int h = lane >> 4;
  v8bf lo = *(const v8bf*)(Bs + n * LDSP + h * 8);
  v8bf hi = *(const v8bf*)(Bs + n * LDSP + 16 + h * 8);
  v16bf r;
#pragma unroll
  for (int i = 0; i < 8; ++i) { r[i] = lo[i]; r[i + 8] = hi[i]; }
  return r;
}

// -------- WcT[n][k] = 0.9*(k==n) - sum_j Aw[j][k]*Aw[j][n] + Bw[n][k] - Bw[k][n]
__global__ void __launch_bounds__(256) mon_wct(const float* __restrict__ Aw,
                                               const float* __restrict__ Bw,
                                               unsigned short* __restrict__ WcT) {
  const int n = blockIdx.x;
  const int k0 = threadIdx.x;
  float dot[4] = {0.f, 0.f, 0.f, 0.f};
  for (int j = 0; j < OUT_DIM; ++j) {
    const float an = Aw[(size_t)j * OUT_DIM + n];
    const float* row = Aw + (size_t)j * OUT_DIM;
#pragma unroll
    for (int i = 0; i < 4; ++i) dot[i] += an * row[k0 + i * 256];
  }
#pragma unroll
  for (int i = 0; i < 4; ++i) {
    const int k = k0 + i * 256;
    float w = ((k == n) ? 0.9f : 0.0f) - dot[i]
              + Bw[(size_t)n * OUT_DIM + k] - Bw[(size_t)k * OUT_DIM + n];
    WcT[(size_t)n * OUT_DIM + k] = f32_bf16(w);
  }
}

// -------- bias = x @ Uw^T + Ub  (f32 in, convert to bf16 while staging to LDS)
//          also emits z1 = ReLU(bias) as bf16
__global__ void __launch_bounds__(256) mon_bias_gemm(const float* __restrict__ X,
                                                     const float* __restrict__ Uw,
                                                     const float* __restrict__ Ub,
                                                     float* __restrict__ biasOut,
                                                     unsigned short* __restrict__ z1) {
  __shared__ unsigned short As[2][BM * LDSP];
  __shared__ unsigned short Bs[2][BN * LDSP];

  const int t = threadIdx.x;
  const int lane = t & 31, wave = t >> 5;
  const int wm = wave >> 2, wn = wave & 3;          // 2 x 4 waves
  const int bm0 = blockIdx.y * BM;
  const int bn0 = blockIdx.x * BN;

  const int r = t >> 1;                              // staging row 0..127
  const int cb = (t & 1) * 16;                       // staging col base

  v8f zero = {};
  v8f acc[4][2];
#pragma unroll
  for (int i = 0; i < 4; ++i)
#pragma unroll
    for (int j = 0; j < 2; ++j) acc[i][j] = zero;

  auto stage = [&](int kt, int buf) {
    const float* ga = X + (size_t)(bm0 + r) * IN_DIM + kt * BK + cb;
    unsigned int* sa = (unsigned int*)&As[buf][r * LDSP + cb];
    const float* gb = Uw + (size_t)(bn0 + r) * IN_DIM + kt * BK + cb;
    unsigned int* sb = (unsigned int*)&Bs[buf][r * LDSP + cb];
#pragma unroll
    for (int i = 0; i < 4; ++i) {
      float4 va = ((const float4*)ga)[i];
      sa[2 * i]     = (unsigned int)f32_bf16(va.x) | ((unsigned int)f32_bf16(va.y) << 16);
      sa[2 * i + 1] = (unsigned int)f32_bf16(va.z) | ((unsigned int)f32_bf16(va.w) << 16);
      float4 vb = ((const float4*)gb)[i];
      sb[2 * i]     = (unsigned int)f32_bf16(vb.x) | ((unsigned int)f32_bf16(vb.y) << 16);
      sb[2 * i + 1] = (unsigned int)f32_bf16(vb.z) | ((unsigned int)f32_bf16(vb.w) << 16);
    }
  };

  const int KT = IN_DIM / BK;  // 96
  stage(0, 0);
  __syncthreads();
  for (int kt = 0; kt < KT; ++kt) {
    if (kt + 1 < KT) stage(kt + 1, (kt + 1) & 1);
    const unsigned short* Ab = As[kt & 1];
    const unsigned short* Bb = Bs[kt & 1];
    v16bf bfr[2];
#pragma unroll
    for (int tn = 0; tn < 2; ++tn) bfr[tn] = load_frag_b(Bb, wn * 32 + tn * 16, lane);
#pragma unroll
    for (int tm = 0; tm < 4; ++tm) {
      v16bf afr = load_frag_a(Ab, wm * 64 + tm * 16, lane);
#pragma unroll
      for (int tn = 0; tn < 2; ++tn)
        acc[tm][tn] = __builtin_amdgcn_wmma_f32_16x16x32_bf16(
            false, afr, false, bfr[tn], (short)0, acc[tm][tn], false, false);
    }
    __syncthreads();
  }

  const int h = lane >> 4;
  const int coln = lane & 15;
#pragma unroll
  for (int tm = 0; tm < 4; ++tm)
#pragma unroll
    for (int tn = 0; tn < 2; ++tn) {
      const int c = bn0 + wn * 32 + tn * 16 + coln;
      const float ub = Ub[c];
#pragma unroll
      for (int v = 0; v < 8; ++v) {
        const int row = bm0 + wm * 64 + tm * 16 + h * 8 + v;
        const float val = acc[tm][tn][v] + ub;
        biasOut[(size_t)row * OUT_DIM + c] = val;
        z1[(size_t)row * OUT_DIM + c] = f32_bf16(fmaxf(val, 0.f));
      }
    }
}

// -------- one FB iteration: zOut = ReLU(zIn @ Wc + bias); last launch writes f32 out
__global__ void __launch_bounds__(256) mon_step_gemm(const unsigned short* __restrict__ Zin,
                                                     const unsigned short* __restrict__ WcT,
                                                     const float* __restrict__ biasIn,
                                                     unsigned short* __restrict__ Zout,
                                                     float* __restrict__ fOut,
                                                     int lastFlag) {
  __shared__ unsigned short As[2][BM * LDSP];
  __shared__ unsigned short Bs[2][BN * LDSP];

  const int t = threadIdx.x;
  const int lane = t & 31, wave = t >> 5;
  const int wm = wave >> 2, wn = wave & 3;
  const int bm0 = blockIdx.y * BM;
  const int bn0 = blockIdx.x * BN;

  const int r = t >> 1;
  const int cb = (t & 1) * 16;

  v8f zero = {};
  v8f acc[4][2];
#pragma unroll
  for (int i = 0; i < 4; ++i)
#pragma unroll
    for (int j = 0; j < 2; ++j) acc[i][j] = zero;

  // Stage one 128x32 bf16 A tile + 128x32 bf16 B tile (N-major) into LDS.
  // Each thread moves 2x16B of A and 2x16B of B.
  auto stage = [&](int kt, int buf) {
    const unsigned short* gA = Zin + (size_t)(bm0 + r) * OUT_DIM + kt * BK + cb;
    unsigned short* lA = &As[buf][r * LDSP + cb];
    const unsigned short* gB = WcT + (size_t)(bn0 + r) * OUT_DIM + kt * BK + cb;
    unsigned short* lB = &Bs[buf][r * LDSP + cb];
#if USE_ASYNC_LDS
    // INST_OFFSET applies to both global and LDS addresses (ISA 10.7/15.18),
    // so offsets 0 and 16 cover the thread's 32-byte slice of each tile.
    __builtin_amdgcn_global_load_async_to_lds_b128((GV4*)gA, (LV4*)lA, 0, 0);
    __builtin_amdgcn_global_load_async_to_lds_b128((GV4*)gA, (LV4*)lA, 16, 0);
    __builtin_amdgcn_global_load_async_to_lds_b128((GV4*)gB, (LV4*)lB, 0, 0);
    __builtin_amdgcn_global_load_async_to_lds_b128((GV4*)gB, (LV4*)lB, 16, 0);
#else
    ((uint4*)lA)[0] = ((const uint4*)gA)[0];
    ((uint4*)lA)[1] = ((const uint4*)gA)[1];
    ((uint4*)lB)[0] = ((const uint4*)gB)[0];
    ((uint4*)lB)[1] = ((const uint4*)gB)[1];
#endif
  };

  const int KT = OUT_DIM / BK;  // 32
  stage(0, 0);
#if USE_ASYNC_LDS
  __builtin_amdgcn_s_wait_asynccnt(0);
#endif
  __syncthreads();
  for (int kt = 0; kt < KT; ++kt) {
    if (kt + 1 < KT) stage(kt + 1, (kt + 1) & 1);  // async copy overlaps WMMAs below
    const unsigned short* Ab = As[kt & 1];
    const unsigned short* Bb = Bs[kt & 1];
    v16bf bfr[2];
#pragma unroll
    for (int tn = 0; tn < 2; ++tn) bfr[tn] = load_frag_b(Bb, wn * 32 + tn * 16, lane);
#pragma unroll
    for (int tm = 0; tm < 4; ++tm) {
      v16bf afr = load_frag_a(Ab, wm * 64 + tm * 16, lane);
#pragma unroll
      for (int tn = 0; tn < 2; ++tn)
        acc[tm][tn] = __builtin_amdgcn_wmma_f32_16x16x32_bf16(
            false, afr, false, bfr[tn], (short)0, acc[tm][tn], false, false);
    }
#if USE_ASYNC_LDS
    __builtin_amdgcn_s_wait_asynccnt(0);  // this wave's prefetch landed in LDS
#endif
    __syncthreads();
  }

  const int h = lane >> 4;
  const int coln = lane & 15;
#pragma unroll
  for (int tm = 0; tm < 4; ++tm)
#pragma unroll
    for (int tn = 0; tn < 2; ++tn) {
      const int c = bn0 + wn * 32 + tn * 16 + coln;
#pragma unroll
      for (int v = 0; v < 8; ++v) {
        const int row = bm0 + wm * 64 + tm * 16 + h * 8 + v;
        float val = acc[tm][tn][v] + biasIn[(size_t)row * OUT_DIM + c];
        val = fmaxf(val, 0.f);
        Zout[(size_t)row * OUT_DIM + c] = f32_bf16(val);
        if (lastFlag) fOut[(size_t)row * OUT_DIM + c] = val;
      }
    }
}

extern "C" void kernel_launch(void* const* d_in, const int* in_sizes, int n_in,
                              void* d_out, int out_size, void* d_ws, size_t ws_size,
                              hipStream_t stream) {
  (void)in_sizes; (void)n_in; (void)out_size; (void)ws_size;
  const float* x  = (const float*)d_in[0];
  const float* Uw = (const float*)d_in[1];
  const float* Ub = (const float*)d_in[2];
  const float* Aw = (const float*)d_in[3];
  const float* Bw = (const float*)d_in[4];
  float* out = (float*)d_out;

  // workspace layout (~69 MB): bias f32 | WcT bf16 | zA bf16 | zB bf16
  char* ws = (char*)d_ws;
  float* bias = (float*)ws;
  size_t off = (size_t)BATCH * OUT_DIM * sizeof(float);
  unsigned short* WcT = (unsigned short*)(ws + off);
  off += (size_t)OUT_DIM * OUT_DIM * sizeof(unsigned short);
  unsigned short* zA = (unsigned short*)(ws + off);
  off += (size_t)BATCH * OUT_DIM * sizeof(unsigned short);
  unsigned short* zB = (unsigned short*)(ws + off);

  mon_wct<<<OUT_DIM, 256, 0, stream>>>(Aw, Bw, WcT);

  dim3 grid(OUT_DIM / BN, BATCH / BM);  // 8 x 64
  mon_bias_gemm<<<grid, 256, 0, stream>>>(x, Uw, Ub, bias, zA);

  unsigned short* zin = zA;
  unsigned short* zout = zB;
  for (int it = 0; it < 50; ++it) {  // z1 done; 50 more -> f^51(0)
    const int last = (it == 49) ? 1 : 0;
    mon_step_gemm<<<grid, 256, 0, stream>>>(zin, WcT, bias, zout, out, last);
    unsigned short* tmp = zin; zin = zout; zout = tmp;
  }
}
